// LSTMStaticConditioning_56487409877455
// MI455X (gfx1250) — matrix operand order repaired and checked by reading, not compiled, not run on hardware
//
#include <hip/hip_runtime.h>

// ---------------- types ----------------
typedef __attribute__((ext_vector_type(16))) __bf16 v16bf;
typedef __attribute__((ext_vector_type(8)))  __bf16 v8bf;
typedef __attribute__((ext_vector_type(8)))  float  v8f;

#define B_   128
#define H_   512
#define G4H  2048
#define TCTX 365
#define TPRD 90
#define TTOT 455
#define EMB_ 64
#define KPAD 96      // 10 dyn + 64 emb padded to 96 (multiple of 32)
#define BH   (B_ * H_)

__device__ __forceinline__ float sigm(float x) { return 1.0f / (1.0f + __expf(-x)); }

// ---------------- small fp32 helpers ----------------
__global__ void cvt_f32_bf16(__bf16* __restrict__ dst, const float* __restrict__ src, int n) {
    int i = blockIdx.x * blockDim.x + threadIdx.x;
    if (i < n) dst[i] = (__bf16)src[i];
}

// dst (drows x dcols) bf16, zero-padded from src (srows x scols) fp32
__global__ void pack_pad_bf16(__bf16* __restrict__ dst, const float* __restrict__ src,
                              int drows, int dcols, int srows, int scols) {
    int i = blockIdx.x * blockDim.x + threadIdx.x;
    int n = drows * dcols;
    if (i >= n) return;
    int r = i / dcols, c = i % dcols;
    float v = (r < srows && c < scols) ? src[r * scols + c] : 0.0f;
    dst[i] = (__bf16)v;
}

__global__ void add_vec(float* __restrict__ dst, const float* __restrict__ a,
                        const float* __restrict__ b, int n) {
    int i = blockIdx.x * blockDim.x + threadIdx.x;
    if (i < n) dst[i] = a[i] + b[i];
}

// ---------------- static conditioning: MLP init state + embedding ----------------
// one block per batch element, 256 threads
__global__ void static_prep(const float* __restrict__ sx_all,
                            const float* __restrict__ w1, const float* __restrict__ b1,
                            const float* __restrict__ w2, const float* __restrict__ b2,
                            const float* __restrict__ ew, const float* __restrict__ eb,
                            __bf16* __restrict__ h0l0, __bf16* __restrict__ h0l1,
                            float* __restrict__ cl0, float* __restrict__ cl1,
                            __bf16* __restrict__ emb) {
    __shared__ float sx[36];
    __shared__ float hid[128];
    int b = blockIdx.x, tid = threadIdx.x;
    if (tid < 36) sx[tid] = sx_all[b * 36 + tid];
    __syncthreads();
    if (tid < 128) {
        float a = b1[tid];
        #pragma unroll 4
        for (int k = 0; k < 36; ++k) a += sx[k] * w1[tid * 36 + k];
        hid[tid] = a > 0.0f ? a : 0.0f;
    }
    __syncthreads();
    // 2048 outputs = reshape(b, 2, L, H): [0]=h0, [1]=c0 ; layer ; hidden
    for (int j = 0; j < 8; ++j) {
        int o = tid * 8 + j;
        float a = b2[o];
        #pragma unroll 4
        for (int k = 0; k < 128; ++k) a += hid[k] * w2[o * 128 + k];
        int s  = o >> 10;        // 0 -> h0, 1 -> c0
        int l  = (o >> 9) & 1;   // layer
        int hh = o & 511;
        if (s == 0) { (l == 0 ? h0l0 : h0l1)[b * H_ + hh] = (__bf16)a; }
        else        { (l == 0 ? cl0  : cl1 )[b * H_ + hh] = a; }
    }
    if (tid < EMB_) {
        float a = eb[tid];
        #pragma unroll 4
        for (int k = 0; k < 36; ++k) a += sx[k] * ew[tid * 36 + k];
        emb[b * EMB_ + tid] = (__bf16)(a > 0.0f ? a : 0.0f);
    }
}

// ---------------- pack [dyn(10) | emb(64) | 0(22)] -> bf16 (TTOT*B, 96) ----------------
__global__ void pack_inputs(__bf16* __restrict__ Xp,
                            const float* __restrict__ ctx,   // (B, TCTX, 10)
                            const float* __restrict__ fut,   // (B, TPRD, 10)
                            const __bf16* __restrict__ emb)  // (B, 64)
{
    int i = blockIdx.x * blockDim.x + threadIdx.x;
    const int n = TTOT * B_ * KPAD;
    if (i >= n) return;
    int c = i % KPAD;
    int r = i / KPAD;      // r = t*B + b
    int b = r % B_;
    int t = r / B_;
    if (c < 10) {
        float v = (t < TCTX) ? ctx[((size_t)b * TCTX + t) * 10 + c]
                             : fut[((size_t)b * TPRD + (t - TCTX)) * 10 + c];
        Xp[i] = (__bf16)v;
    } else if (c < 74) {
        Xp[i] = emb[b * EMB_ + (c - 10)];
    } else {
        Xp[i] = (__bf16)0.0f;
    }
}

// ---------------- WMMA bf16 GEMM: C = A (MxK) * Bt^T (N x K) + bias ----------------
// one wave per 16x64 output strip (4 tiles sharing the A fragment);
// K multiple of 32; M multiple of 16; N multiple of 64
__global__ void wmma_gemm_bf16(const __bf16* __restrict__ A, int lda,
                               const __bf16* __restrict__ Bt, int ldb,
                               const float* __restrict__ bias,
                               void* __restrict__ out, int ldc,
                               int M, int N, int K, int nreal, int flags /*1=relu 2=bf16out*/) {
    int wave = (blockIdx.x * blockDim.x + threadIdx.x) >> 5;
    int nStrips = N >> 6;
    int tot = (M >> 4) * nStrips;
    if (wave >= tot) return;
    int mT = wave / nStrips, sT = wave % nStrips;
    int m0 = mT << 4, n0 = sT << 6;

    int lane  = threadIdx.x & 31;
    int lhalf = lane >> 4;     // 0 or 1
    int l16   = lane & 15;

    const __bf16* Arow = A + (size_t)(m0 + l16) * lda;
    const __bf16* Br0 = Bt + (size_t)(n0 +  0 + l16) * ldb;
    const __bf16* Br1 = Bt + (size_t)(n0 + 16 + l16) * ldb;
    const __bf16* Br2 = Bt + (size_t)(n0 + 32 + l16) * ldb;
    const __bf16* Br3 = Bt + (size_t)(n0 + 48 + l16) * ldb;

    v8f acc0 = {}, acc1 = {}, acc2 = {}, acc3 = {};
    for (int k0 = 0; k0 < K; k0 += 32) {
        int ca = k0 + lhalf * 16;
        v8bf alo = *(const v8bf*)(Arow + ca);
        v8bf ahi = *(const v8bf*)(Arow + ca + 8);
        v16bf a = __builtin_shufflevector(alo, ahi, 0,1,2,3,4,5,6,7,8,9,10,11,12,13,14,15);
        int cb = k0 + lhalf * 16;
        v16bf b0 = *(const v16bf*)(Br0 + cb);
        v16bf b1 = *(const v16bf*)(Br1 + cb);
        v16bf b2 = *(const v16bf*)(Br2 + cb);
        v16bf b3 = *(const v16bf*)(Br3 + cb);
        acc0 = __builtin_amdgcn_wmma_f32_16x16x32_bf16(false, a, false, b0, (short)0, acc0, false, false);
        acc1 = __builtin_amdgcn_wmma_f32_16x16x32_bf16(false, a, false, b1, (short)0, acc1, false, false);
        acc2 = __builtin_amdgcn_wmma_f32_16x16x32_bf16(false, a, false, b2, (short)0, acc2, false, false);
        acc3 = __builtin_amdgcn_wmma_f32_16x16x32_bf16(false, a, false, b3, (short)0, acc3, false, false);
    }

    #pragma unroll
    for (int j = 0; j < 4; ++j) {
        v8f acc = (j == 0) ? acc0 : (j == 1) ? acc1 : (j == 2) ? acc2 : acc3;
        int n = n0 + j * 16 + l16;
        if (n >= nreal) continue;
        float bv = bias ? bias[n] : 0.0f;
        #pragma unroll
        for (int r = 0; r < 8; ++r) {
            int m = m0 + r + lhalf * 8;
            float v = acc[r] + bv;
            if (flags & 1) v = v > 0.0f ? v : 0.0f;
            if (flags & 2) ((__bf16*)out)[(size_t)m * ldc + n] = (__bf16)v;
            else           ((float*)out) [(size_t)m * ldc + n] = v;
        }
    }
}

// ---------------- one LSTM time step: g = xg + h*Whh^T ; gates ; update c,h ----------------
// grid: 64 blocks x 128 threads = 256 waves; wave (mT, jT) owns all 4 gates of tile
__global__ void __launch_bounds__(128)
lstm_step(const float* __restrict__ xg,      // (B, 4H) for this t
          const __bf16* __restrict__ h_in,   // (B, H)
          float* __restrict__ c,             // (B, H) in-place
          __bf16* __restrict__ h_out,        // (B, H)
          const __bf16* __restrict__ whh)    // (4H, H) row-major
{
    int wave = (blockIdx.x * blockDim.x + threadIdx.x) >> 5;
    int mT = wave >> 5;          // 0..7   (B/16)
    int jT = wave & 31;          // 0..31  (H/16)
    int m0 = mT << 4, j0 = jT << 4;

    int lane  = threadIdx.x & 31;
    int lhalf = lane >> 4;
    int l16   = lane & 15;

    // init accumulators from precomputed input contribution (torch gate order i,f,g,o)
    v8f ai = {}, af = {}, ag = {}, ao = {};
    #pragma unroll
    for (int r = 0; r < 8; ++r) {
        int m = m0 + r + lhalf * 8;
        const float* xr = xg + (size_t)m * G4H + j0 + l16;
        ai[r] = xr[0];
        af[r] = xr[512];
        ag[r] = xr[1024];
        ao[r] = xr[1536];
    }

    const __bf16* Arow = h_in + (size_t)(m0 + l16) * H_;
    const __bf16* B0 = whh + (size_t)(   0 + j0 + l16) * H_;
    const __bf16* B1 = whh + (size_t)( 512 + j0 + l16) * H_;
    const __bf16* B2 = whh + (size_t)(1024 + j0 + l16) * H_;
    const __bf16* B3 = whh + (size_t)(1536 + j0 + l16) * H_;

    for (int k0 = 0; k0 < H_; k0 += 32) {
        int ca = k0 + lhalf * 16;
        v8bf alo = *(const v8bf*)(Arow + ca);
        v8bf ahi = *(const v8bf*)(Arow + ca + 8);
        v16bf a = __builtin_shufflevector(alo, ahi, 0,1,2,3,4,5,6,7,8,9,10,11,12,13,14,15);
        int cb = k0 + lhalf * 16;
        v16bf b0 = *(const v16bf*)(B0 + cb);
        v16bf b1 = *(const v16bf*)(B1 + cb);
        v16bf b2 = *(const v16bf*)(B2 + cb);
        v16bf b3 = *(const v16bf*)(B3 + cb);
        ai = __builtin_amdgcn_wmma_f32_16x16x32_bf16(false, a, false, b0, (short)0, ai, false, false);
        af = __builtin_amdgcn_wmma_f32_16x16x32_bf16(false, a, false, b1, (short)0, af, false, false);
        ag = __builtin_amdgcn_wmma_f32_16x16x32_bf16(false, a, false, b2, (short)0, ag, false, false);
        ao = __builtin_amdgcn_wmma_f32_16x16x32_bf16(false, a, false, b3, (short)0, ao, false, false);
    }

    #pragma unroll
    for (int r = 0; r < 8; ++r) {
        int m = m0 + r + lhalf * 8;
        int n = j0 + l16;
        size_t ci = (size_t)m * H_ + n;
        float iv = sigm(ai[r]);
        float fv = sigm(af[r]);
        float gv = tanhf(ag[r]);
        float ov = sigm(ao[r]);
        float cn = fv * c[ci] + iv * gv;
        c[ci] = cn;
        h_out[ci] = (__bf16)(ov * tanhf(cn));
    }
}

// ---------------- final head: preds = hid(11520x256 bf16) @ head_w2^T + b2, permuted ----------------
__global__ void head_out(const __bf16* __restrict__ hid, const float* __restrict__ w2,
                         const float* __restrict__ b2, float* __restrict__ out) {
    int r = blockIdx.x * blockDim.x + threadIdx.x;   // r = tp*B + b
    if (r >= TPRD * B_) return;
    int tp = r / B_, b = r % B_;
    float a0 = b2[0], a1 = b2[1], a2 = b2[2], a3 = b2[3];
    const __bf16* hr = hid + (size_t)r * 256;
    #pragma unroll 4
    for (int k = 0; k < 256; ++k) {
        float hv = (float)hr[k];
        a0 += hv * w2[0 * 256 + k];
        a1 += hv * w2[1 * 256 + k];
        a2 += hv * w2[2 * 256 + k];
        a3 += hv * w2[3 * 256 + k];
    }
    float* op = out + (size_t)b * (TPRD * 4) + (size_t)tp * 4;
    op[0] = a0; op[1] = a1; op[2] = a2; op[3] = a3;
}

// ---------------- host ----------------
extern "C" void kernel_launch(void* const* d_in, const int* in_sizes, int n_in,
                              void* d_out, int out_size, void* d_ws, size_t ws_size,
                              hipStream_t stream) {
    const float* context_x = (const float*)d_in[0];
    const float* static_x  = (const float*)d_in[1];
    const float* future_x  = (const float*)d_in[2];
    const float* mlp_w1 = (const float*)d_in[3];
    const float* mlp_b1 = (const float*)d_in[4];
    const float* mlp_w2 = (const float*)d_in[5];
    const float* mlp_b2 = (const float*)d_in[6];
    const float* emb_w  = (const float*)d_in[7];
    const float* emb_b  = (const float*)d_in[8];
    const float* w_ih0  = (const float*)d_in[9];
    const float* w_hh0  = (const float*)d_in[10];
    const float* b_ih0  = (const float*)d_in[11];
    const float* b_hh0  = (const float*)d_in[12];
    const float* w_ih1  = (const float*)d_in[13];
    const float* w_hh1  = (const float*)d_in[14];
    const float* b_ih1  = (const float*)d_in[15];
    const float* b_hh1  = (const float*)d_in[16];
    const float* head_w1 = (const float*)d_in[17];
    const float* head_b1 = (const float*)d_in[18];
    const float* head_w2 = (const float*)d_in[19];
    const float* head_b2 = (const float*)d_in[20];
    float* out = (float*)d_out;

    char* base = (char*)d_ws;
    size_t off = 0;
    auto alloc = [&](size_t bytes) -> void* {
        void* p = base + off;
        off = (off + bytes + 255) & ~(size_t)255;
        return p;
    };

    __bf16* whh0b = (__bf16*)alloc((size_t)G4H * H_ * 2);
    __bf16* whh1b = (__bf16*)alloc((size_t)G4H * H_ * 2);
    __bf16* wih1b = (__bf16*)alloc((size_t)G4H * H_ * 2);
    __bf16* wp0   = (__bf16*)alloc((size_t)G4H * KPAD * 2);
    __bf16* hw1b  = (__bf16*)alloc((size_t)256 * H_ * 2);
    float*  bias0 = (float*) alloc((size_t)G4H * 4);
    float*  bias1 = (float*) alloc((size_t)G4H * 4);
    __bf16* emb   = (__bf16*)alloc((size_t)B_ * EMB_ * 2);
    __bf16* h0l0  = (__bf16*)alloc((size_t)BH * 2);
    __bf16* h0l1  = (__bf16*)alloc((size_t)BH * 2);
    float*  cl0   = (float*) alloc((size_t)BH * 4);
    float*  cl1   = (float*) alloc((size_t)BH * 4);
    __bf16* Xp    = (__bf16*)alloc((size_t)TTOT * B_ * KPAD * 2);
    float*  xg0   = (float*) alloc((size_t)TTOT * B_ * G4H * 4);
    float*  xg1   = (float*) alloc((size_t)TTOT * B_ * G4H * 4);
    __bf16* ys0   = (__bf16*)alloc((size_t)TTOT * BH * 2);
    __bf16* ys1   = (__bf16*)alloc((size_t)TTOT * BH * 2);
    __bf16* hid   = (__bf16*)alloc((size_t)TPRD * B_ * 256 * 2);

    auto cdiv = [](int a, int b) { return (a + b - 1) / b; };

    // 1) weight conversion / packing
    cvt_f32_bf16<<<cdiv(G4H * H_, 256), 256, 0, stream>>>(whh0b, w_hh0, G4H * H_);
    cvt_f32_bf16<<<cdiv(G4H * H_, 256), 256, 0, stream>>>(whh1b, w_hh1, G4H * H_);
    cvt_f32_bf16<<<cdiv(G4H * H_, 256), 256, 0, stream>>>(wih1b, w_ih1, G4H * H_);
    cvt_f32_bf16<<<cdiv(256 * H_, 256), 256, 0, stream>>>(hw1b, head_w1, 256 * H_);
    pack_pad_bf16<<<cdiv(G4H * KPAD, 256), 256, 0, stream>>>(wp0, w_ih0, G4H, KPAD, G4H, 74);
    add_vec<<<cdiv(G4H, 256), 256, 0, stream>>>(bias0, b_ih0, b_hh0, G4H);
    add_vec<<<cdiv(G4H, 256), 256, 0, stream>>>(bias1, b_ih1, b_hh1, G4H);

    // 2) static conditioning
    static_prep<<<B_, 256, 0, stream>>>(static_x, mlp_w1, mlp_b1, mlp_w2, mlp_b2,
                                        emb_w, emb_b, h0l0, h0l1, cl0, cl1, emb);

    // 3) pack timestep inputs
    pack_inputs<<<cdiv(TTOT * B_ * KPAD, 256), 256, 0, stream>>>(Xp, context_x, future_x, emb);

    auto gemm = [&](const __bf16* A, int lda, const __bf16* Bt, int ldb, const float* bias,
                    void* o, int ldc, int M, int N, int K, int nreal, int flags) {
        int strips = (M >> 4) * (N >> 6);
        wmma_gemm_bf16<<<cdiv(strips, 4), 128, 0, stream>>>(A, lda, Bt, ldb, bias, o, ldc,
                                                            M, N, K, nreal, flags);
    };

    const int Mall = TTOT * B_;

    // 4) layer-0 input contribution for all timesteps
    gemm(Xp, KPAD, wp0, KPAD, bias0, xg0, G4H, Mall, G4H, KPAD, G4H, 0);

    // 5) layer-0 recurrence (writes h directly into ys0[t])
    for (int t = 0; t < TTOT; ++t) {
        const __bf16* hin = (t == 0) ? h0l0 : (ys0 + (size_t)(t - 1) * BH);
        lstm_step<<<64, 128, 0, stream>>>(xg0 + (size_t)t * B_ * G4H, hin, cl0,
                                          ys0 + (size_t)t * BH, whh0b);
    }

    // 6) layer-1 input contribution for all timesteps
    gemm(ys0, H_, wih1b, H_, bias1, xg1, G4H, Mall, G4H, H_, G4H, 0);

    // 7) layer-1 recurrence
    for (int t = 0; t < TTOT; ++t) {
        const __bf16* hin = (t == 0) ? h0l1 : (ys1 + (size_t)(t - 1) * BH);
        lstm_step<<<64, 128, 0, stream>>>(xg1 + (size_t)t * B_ * G4H, hin, cl1,
                                          ys1 + (size_t)t * BH, whh1b);
    }

    // 8) output head on prediction steps: hid = relu(ys1[365:] @ head_w1^T + b1)
    const __bf16* ys1fut = ys1 + (size_t)TCTX * BH;
    gemm(ys1fut, H_, hw1b, H_, head_b1, hid, 256, TPRD * B_, 256, H_, 256, 1 | 2);

    // 9) final projection + (b, tp, out) permutation
    head_out<<<cdiv(TPRD * B_, 256), 256, 0, stream>>>(hid, head_w2, head_b2, out);
}
